// ResidualQuantizer_83373905150270
// MI455X (gfx1250) — matrix operand order repaired and checked by reading, not compiled, not run on hardware
//
#include <hip/hip_runtime.h>
#include <hip/hip_bf16.h>
#include <cstdint>
#include <cstddef>

// Problem constants (match reference)
#define LSTAGES 4
#define KCODES  1024
#define DIM     256
#define BATCH   65536
#define BETA_C    0.25f
#define ENT_REG_C 1e-4f
#define DECAY_C   0.99f
#define EPS_C     1e-5f

typedef __attribute__((ext_vector_type(16))) __bf16 v16bf;
typedef __attribute__((ext_vector_type(8)))  float  v8f;
typedef __attribute__((ext_vector_type(4)))  unsigned int u32x4;
typedef __attribute__((ext_vector_type(8)))  int i32x8;
typedef __attribute__((ext_vector_type(4)))  int i32x4;

union FragBF {
    v16bf v;
    uint4 u[2];
};

__device__ __forceinline__ unsigned short f32_to_bf16(float f) {
    union { float f; unsigned int u; } x;
    x.f = f;
    unsigned int u = x.u;
    unsigned int r = u + 0x7FFFu + ((u >> 16) & 1u);   // round-to-nearest-even
    return (unsigned short)(r >> 16);
}

#if defined(__has_builtin)
#if __has_builtin(__builtin_amdgcn_tensor_load_to_lds)
#define HAVE_TDM 1
#endif
#endif

#ifdef HAVE_TDM
// One TDM descriptor: contiguous copy of `nbytes` (multiple of 8) from global
// `gaddr` into LDS byte offset `lds_off`, expressed as a 1D tile of 8-byte
// elements (D# layout per CDNA5 ISA 8.3/8.4; type=2 "image", count=1).
__device__ __forceinline__ void tdm_copy_to_lds(unsigned lds_off, const void* gaddr,
                                                unsigned nbytes) {
    const unsigned long long ga = (unsigned long long)(uintptr_t)gaddr;
    const unsigned elems = nbytes >> 3;          // 8-byte data_size units
    u32x4 g0;
    g0[0] = 1u;                                              // count=1 (valid user D#)
    g0[1] = lds_off;                                         // lds_addr (bytes)
    g0[2] = (unsigned)(ga & 0xFFFFFFFFu);                    // global_addr[31:0]
    g0[3] = (unsigned)((ga >> 32) & 0x1FFFFFFu) | (2u << 30); // global_addr[56:32] | type=2
    i32x8 g1;
    g1[0] = (int)(3u << 16);                                 // data_size=3 (8B), no flags
    g1[1] = (int)((elems & 0xFFFFu) << 16);                  // tensor_dim0[15:0]
    g1[2] = (int)(((elems >> 16) & 0xFFFFu) | (1u << 16));   // tensor_dim0[31:16], tensor_dim1=1
    g1[3] = (int)((elems & 0xFFFFu) << 16);                  // tile_dim0 = elems
    g1[4] = 1;                                               // tile_dim1=1, tile_dim2=0
    g1[5] = (int)elems;                                      // tensor_dim0_stride[31:0]
    g1[6] = 0;
    g1[7] = 0;
    i32x4 z4 = {0, 0, 0, 0};
#if defined(__clang_major__) && (__clang_major__ >= 23)
    i32x8 z8 = {0, 0, 0, 0, 0, 0, 0, 0};
    __builtin_amdgcn_tensor_load_to_lds(g0, g1, z4, z4, z8, 0);
#else
    __builtin_amdgcn_tensor_load_to_lds(g0, g1, z4, z4, 0);
#endif
}
#endif

// ---------------------------------------------------------------------------
// Kernel 1: convert codebooks to bf16 + compute per-code squared norms.
// grid = L*K blocks, 256 threads (one per dim).
// ---------------------------------------------------------------------------
__global__ void __launch_bounds__(256) rvq_prep(const float* __restrict__ cb,
                                                unsigned short* __restrict__ cbb,
                                                float* __restrict__ cnorm) {
    const int blk = blockIdx.x;            // l*K + k
    const int d   = threadIdx.x;
    float v = cb[(size_t)blk * DIM + d];
    cbb[(size_t)blk * DIM + d] = f32_to_bf16(v);

    float s = v * v;
    __shared__ float red[8];
    #pragma unroll
    for (int o = 16; o > 0; o >>= 1) s += __shfl_xor(s, o, 32);
    if ((threadIdx.x & 31) == 0) red[threadIdx.x >> 5] = s;
    __syncthreads();
    if (threadIdx.x == 0) {
        float t = 0.f;
        #pragma unroll
        for (int w = 0; w < 8; ++w) t += red[w];
        cnorm[blk] = t;
    }
}

// ---------------------------------------------------------------------------
// Kernel 2: WMMA nearest-neighbor search.
// 256 threads = 8 waves; each wave owns 16 rows. Block covers 128 rows.
// grid = BATCH/128 = 512 blocks.
// LDS: 64KB residual(bf16) + 2x32KB ping-pong codebook tiles + 2x256B norms.
// TDM moves tile t+1 while WMMAs consume tile t (TENSORcnt overlap).
// 4 accumulators cover the whole 64-code tile so each dscnt drain feeds
// 4 back-to-back WMMAs sharing one A fragment.
// dist(m,n) = ||C_n||^2 - 2 * dot(R_m, C_n)   (||R||^2 dropped: argmin-invariant)
// ---------------------------------------------------------------------------
__global__ void __launch_bounds__(256) rvq_nn(const float* __restrict__ resid,
                                              const unsigned short* __restrict__ cbb,
                                              const float* __restrict__ cnorm,
                                              int* __restrict__ idx_out,
                                              float* __restrict__ ids_out) {
    extern __shared__ char smem[];
    unsigned short* resid_s  = (unsigned short*)smem;                       // [128][256] bf16
    unsigned short* cb_s     = (unsigned short*)(smem + 128 * DIM * 2);     // [2][64][256] bf16
    float*          cnorm_sb = (float*)(smem + 128 * DIM * 2 + 2 * 64 * DIM * 2); // [2][64]

    const int tid  = threadIdx.x;
    const int lane = tid & 31;
    const int wave = tid >> 5;
    const int half = lane >> 4;    // 0 or 1 (lane group)
    const int lm   = lane & 15;

    // ---- stage residual tile f32 -> bf16 into LDS ----
    const float2* rsrc = (const float2*)(resid + (size_t)blockIdx.x * 128 * DIM);
    unsigned int* rdst = (unsigned int*)resid_s;
    for (int i = tid; i < 128 * DIM / 2; i += 256) {
        float2 f = rsrc[i];
        rdst[i] = (unsigned int)f32_to_bf16(f.x) | ((unsigned int)f32_to_bf16(f.y) << 16);
    }
    __syncthreads();

    // ---- load A fragments (16-bit A 16x32 layout, ISA 7.12.2) ----
    // lane holds row M = lane&15; element e -> K = (e&7) + 8*half + 16*(e>>3)
    FragBF a[8];
    const unsigned short* arow = resid_s + (size_t)(wave * 16 + lm) * DIM;
    #pragma unroll
    for (int dc = 0; dc < 8; ++dc) {
        a[dc].u[0] = *(const uint4*)(arow + dc * 32 + 8 * half);
        a[dc].u[1] = *(const uint4*)(arow + dc * 32 + 16 + 8 * half);
    }

    float best[8];
    int   bidx[8];
    #pragma unroll
    for (int r = 0; r < 8; ++r) { best[r] = 3.4e38f; bidx[r] = 0; }

    const unsigned cb_lds_base = (unsigned)(uintptr_t)(void*)cb_s;
    const int NT = KCODES / 64;   // 16 tiles

#ifdef HAVE_TDM
    // prologue: start DMA of tile 0 into buffer 0
    if (wave == 0)
        tdm_copy_to_lds(cb_lds_base, cbb, 64 * DIM * 2);
#endif

    for (int t = 0; t < NT; ++t) {
        const int buf = t & 1;
#ifdef HAVE_TDM
        if (wave == 0) {
            // tile t is the only outstanding tensor op here -> constant wait 0
            __builtin_amdgcn_s_wait_tensorcnt(0);
            // kick tile t+1 into the other buffer; overlaps with compute below
            if (t + 1 < NT)
                tdm_copy_to_lds(cb_lds_base + (unsigned)((t + 1) & 1) * (64 * DIM * 2),
                                cbb + (size_t)(t + 1) * 64 * DIM, 64 * DIM * 2);
        }
#else
        {
            const uint4* gsrc = (const uint4*)(cbb + (size_t)t * 64 * DIM);
            uint4*       ldst = (uint4*)(cb_s + (size_t)buf * 64 * DIM);
            for (int i = tid; i < 64 * DIM * 2 / 16; i += 256) ldst[i] = gsrc[i];
        }
#endif
        if (tid < 64) cnorm_sb[buf * 64 + tid] = cnorm[t * 64 + tid];
        if (t + 1 < NT)
            __builtin_prefetch(cbb + (size_t)(t + 1) * 64 * DIM, 0, 1);
        __syncthreads();

        const unsigned short* cbuf = cb_s + (size_t)buf * 64 * DIM;

        // 4 accumulators = 4 N-tiles = whole 64-code LDS tile per D-chunk pass
        v8f acc[4];
        #pragma unroll
        for (int j = 0; j < 4; ++j) acc[j] = (v8f){0.f,0.f,0.f,0.f,0.f,0.f,0.f,0.f};

        #pragma unroll
        for (int dc = 0; dc < 8; ++dc) {
            // B fragments (16-bit B 32x16): lane -> column N = lane&15,
            // elements -> K = (e&7) + 8*half + 16*(e>>3) (same as A)
            FragBF b[4];
            #pragma unroll
            for (int j = 0; j < 4; ++j) {
                const unsigned short* brow = cbuf + (size_t)(j * 16 + lm) * DIM;
                const uint4* bp = (const uint4*)(brow + dc * 32 + 8 * half);
                b[j].u[0] = bp[0];
                b[j].u[1] = bp[2];
            }
            #pragma unroll
            for (int j = 0; j < 4; ++j) {
                acc[j] = __builtin_amdgcn_wmma_f32_16x16x32_bf16(
                    false, a[dc].v, false, b[j].v, (short)0, acc[j], false, false);
            }
        }

        // C/D layout: lane (l%16) = column, reg r = row (M = r + 8*(l/16))
        #pragma unroll
        for (int j = 0; j < 4; ++j) {
            float cn    = cnorm_sb[buf * 64 + j * 16 + lm];
            int   gcode = t * 64 + j * 16 + lm;
            #pragma unroll
            for (int r = 0; r < 8; ++r) {
                float dist = __builtin_fmaf(-2.0f, acc[j][r], cn);
                if (dist < best[r]) { best[r] = dist; bidx[r] = gcode; }
            }
        }
        __syncthreads();   // tile consumed before TDM of t+2 overwrites this buffer
    }

    // ---- cross-lane argmin within each 16-lane half (columns of one row set) ----
    #pragma unroll
    for (int o = 1; o < 16; o <<= 1) {
        #pragma unroll
        for (int r = 0; r < 8; ++r) {
            float ov = __shfl_xor(best[r], o, 32);
            int   oi = __shfl_xor(bidx[r], o, 32);
            if (ov < best[r] || (ov == best[r] && oi < bidx[r])) {
                best[r] = ov; bidx[r] = oi;
            }
        }
    }
    if (lm == 0) {
        const int rowbase = blockIdx.x * 128 + wave * 16 + 8 * half;
        #pragma unroll
        for (int r = 0; r < 8; ++r) {
            idx_out[rowbase + r] = bidx[r];
            ids_out[rowbase + r] = (float)bidx[r];
        }
    }
}

// ---------------------------------------------------------------------------
// Kernel 3: gather q, update residual/zq, scatter counts & dw.
// grid = BATCH blocks, 256 threads (one per dim).
// ---------------------------------------------------------------------------
__global__ void __launch_bounds__(256) rvq_apply(float* __restrict__ resid,
                                                 const float* __restrict__ cb,
                                                 const int* __restrict__ idx,
                                                 float* __restrict__ zq,
                                                 float* __restrict__ dw,
                                                 float* __restrict__ counts) {
    const int b = blockIdx.x;
    const int d = threadIdx.x;
    const int code = idx[b];
    const float q  = cb[(size_t)code * DIM + d];
    const size_t off = (size_t)b * DIM + d;
    const float r = resid[off];
    atomicAdd(&dw[(size_t)code * DIM + d], r);   // segment_sum of PRE-update residual
    resid[off] = r - q;
    zq[off] += q;
    if (d == 0) atomicAdd(&counts[code], 1.0f);
}

// ---------------------------------------------------------------------------
// Kernel 4: EMA update + normalized codebook + entropy. 1 block, 1024 threads.
// ---------------------------------------------------------------------------
__device__ __forceinline__ float block_sum_1024(float v) {
    __shared__ float red[32];
    __shared__ float tot;
    const int lane = threadIdx.x & 31, wid = threadIdx.x >> 5;
    #pragma unroll
    for (int o = 16; o > 0; o >>= 1) v += __shfl_xor(v, o, 32);
    if (lane == 0) red[wid] = v;
    __syncthreads();
    if (wid == 0) {
        float t = red[lane];
        #pragma unroll
        for (int o = 16; o > 0; o >>= 1) t += __shfl_xor(t, o, 32);
        if (lane == 0) tot = t;
    }
    __syncthreads();
    return tot;
}

__global__ void __launch_bounds__(1024) rvq_ema(const float* __restrict__ counts,
                                                const float* __restrict__ dw,
                                                const float* __restrict__ ecs_in,
                                                const float* __restrict__ ew_in,
                                                float* __restrict__ cb_out,
                                                float* __restrict__ ecs_out,
                                                float* __restrict__ ew_out,
                                                float* __restrict__ ent_acc) {
    const int k = threadIdx.x;   // 0..1023
    const float cnt = counts[k];
    const float ecs = ecs_in[k] * DECAY_C + (1.0f - DECAY_C) * cnt;
    ecs_out[k] = ecs;

    const float n  = block_sum_1024(ecs);
    const float cs = (ecs + EPS_C) / (n + (float)KCODES * EPS_C) * n;
    const float inv = 1.0f / cs;

    for (int d = 0; d < DIM; ++d) {
        const size_t off = (size_t)k * DIM + d;
        float ew = ew_in[off] * DECAY_C + (1.0f - DECAY_C) * dw[off];
        ew_out[off] = ew;
        cb_out[off] = ew * inv;
    }

    // batch-usage entropy; counts.sum() == BATCH exactly
    const float p    = cnt * (1.0f / (float)BATCH);
    const float term = p * __logf(fmaxf(p, 1e-9f));
    const float tsum = block_sum_1024(term);
    if (threadIdx.x == 0) atomicAdd(ent_acc, -tsum);
}

// ---------------------------------------------------------------------------
// Kernel 5: sum of squared final residual (for commit loss).
// ---------------------------------------------------------------------------
__global__ void __launch_bounds__(256) rvq_commit(const float* __restrict__ resid,
                                                  float* __restrict__ acc) {
    float s = 0.f;
    const size_t total = (size_t)BATCH * DIM;
    for (size_t i = (size_t)blockIdx.x * blockDim.x + threadIdx.x; i < total;
         i += (size_t)gridDim.x * blockDim.x) {
        float r = resid[i];
        s = __builtin_fmaf(r, r, s);
    }
    __shared__ float red[8];
    const int lane = threadIdx.x & 31, wid = threadIdx.x >> 5;
    #pragma unroll
    for (int o = 16; o > 0; o >>= 1) s += __shfl_xor(s, o, 32);
    if (lane == 0) red[wid] = s;
    __syncthreads();
    if (threadIdx.x == 0) {
        float t = 0.f;
        #pragma unroll
        for (int w = 0; w < 8; ++w) t += red[w];
        atomicAdd(acc, t);
    }
}

// ---------------------------------------------------------------------------
// Kernel 6: write scalar losses.
// ---------------------------------------------------------------------------
__global__ void rvq_final(const float* __restrict__ scal, float* __restrict__ out2) {
    if (threadIdx.x == 0 && blockIdx.x == 0) {
        out2[0] = BETA_C * scal[0] / (float)((size_t)BATCH * DIM);   // commit
        out2[1] = -ENT_REG_C * (scal[1] / (float)LSTAGES);           // loss_reg
    }
}

// ---------------------------------------------------------------------------
extern "C" void kernel_launch(void* const* d_in, const int* in_sizes, int n_in,
                              void* d_out, int out_size, void* d_ws, size_t ws_size,
                              hipStream_t stream) {
    (void)in_sizes; (void)n_in; (void)out_size; (void)ws_size;

    const float* z_e       = (const float*)d_in[0];   // [B,D]
    const float* codebooks = (const float*)d_in[1];   // [L,K,D]
    const float* ecs_in    = (const float*)d_in[2];   // [L,K]
    const float* ew_in     = (const float*)d_in[3];   // [L,K,D]
    float* out = (float*)d_out;

    // d_out layout (flat, reference return order)
    const size_t off_zq     = 0;
    const size_t off_ids    = (size_t)BATCH * DIM;
    const size_t off_commit = off_ids + (size_t)LSTAGES * BATCH;
    const size_t off_loss   = off_commit + 1;
    const size_t off_cb     = off_loss + 1;
    const size_t off_ecs    = off_cb + (size_t)LSTAGES * KCODES * DIM;
    const size_t off_ew     = off_ecs + (size_t)LSTAGES * KCODES;

    // workspace layout
    char* ws = (char*)d_ws;
    size_t o = 0;
    float* resid = (float*)(ws + o);           o += (size_t)BATCH * DIM * 4;
    unsigned short* cbb = (unsigned short*)(ws + o); o += (size_t)LSTAGES * KCODES * DIM * 2;
    float* cnorm  = (float*)(ws + o);          o += (size_t)LSTAGES * KCODES * 4;
    float* counts = (float*)(ws + o);          o += (size_t)KCODES * 4;
    float* dw     = (float*)(ws + o);          o += (size_t)KCODES * DIM * 4;
    int*   idxb   = (int*)(ws + o);            o += (size_t)BATCH * 4;
    float* scal   = (float*)(ws + o);          o += 2 * 4;   // [sum_sq, ent_sum]

    hipMemcpyAsync(resid, z_e, (size_t)BATCH * DIM * 4, hipMemcpyDeviceToDevice, stream);
    hipMemsetAsync(out + off_zq, 0, (size_t)BATCH * DIM * 4, stream);
    hipMemsetAsync(scal, 0, 2 * 4, stream);

    rvq_prep<<<LSTAGES * KCODES, 256, 0, stream>>>(codebooks, cbb, cnorm);

    // 64KB residual + 2x32KB codebook ping-pong + 2x64 norms
    const size_t nn_lds = 128 * DIM * 2 + 2 * 64 * DIM * 2 + 2 * 64 * 4;  // 131584 B

    for (int l = 0; l < LSTAGES; ++l) {
        hipMemsetAsync(counts, 0, (size_t)KCODES * 4, stream);
        hipMemsetAsync(dw, 0, (size_t)KCODES * DIM * 4, stream);

        rvq_nn<<<BATCH / 128, 256, nn_lds, stream>>>(
            resid,
            cbb + (size_t)l * KCODES * DIM,
            cnorm + (size_t)l * KCODES,
            idxb,
            out + off_ids + (size_t)l * BATCH);

        rvq_apply<<<BATCH, 256, 0, stream>>>(
            resid,
            codebooks + (size_t)l * KCODES * DIM,
            idxb,
            out + off_zq,
            dw,
            counts);

        rvq_ema<<<1, 1024, 0, stream>>>(
            counts, dw,
            ecs_in + (size_t)l * KCODES,
            ew_in + (size_t)l * KCODES * DIM,
            out + off_cb + (size_t)l * KCODES * DIM,
            out + off_ecs + (size_t)l * KCODES,
            out + off_ew + (size_t)l * KCODES * DIM,
            scal + 1);
    }

    rvq_commit<<<2048, 256, 0, stream>>>(resid, scal);
    rvq_final<<<1, 1, 0, stream>>>(scal, out + off_commit);
}
